// SAGE_50568944943116
// MI455X (gfx1250) — matrix operand order repaired and checked by reading, not compiled, not run on hardware
//
#include <hip/hip_runtime.h>
#include <math.h>

// ============================================================================
// GraphSAGE 3-layer forward for MI455X (gfx1250, wave32).
//
// v3: GEMM uses a 32x64 wave tile. Per 32-wide K chunk: 12 global_load_b128
// (2 A-frags + 4 B-frags, all at immediate offsets from just TWO moving
// pointers) feed 8 v_wmma_f32_16x16x32_bf16 -- B frags are reused across the
// two M-subtiles, and the serialized 64-bit address chain from v2 is gone.
// All f32->bf16 conversion, K/N zero-padding and degree normalization remain
// baked into the pack kernels; the dual GEMM (agg@Wl + x_tgt@Wr) is a single
// pass over the K-concatenated packed operands.
// ============================================================================

typedef __attribute__((ext_vector_type(16))) __bf16 v16bf;
typedef __attribute__((ext_vector_type(8)))  float  v8f;

// ---------------------------------------------------------------------------
// Utility kernels
// ---------------------------------------------------------------------------
__global__ void zero_u32_kernel(unsigned int* __restrict__ p, long long n) {
  long long i = (long long)blockIdx.x * blockDim.x + threadIdx.x;
  if (i < n) p[i] = 0u;
}

__global__ void count_kernel(const int* __restrict__ dst, int* __restrict__ cnt, int E) {
  int i = blockIdx.x * blockDim.x + threadIdx.x;
  if (i < E) atomicAdd(&cnt[dst[i]], 1);
}

__global__ void invdeg_kernel(const int* __restrict__ cnt, float* __restrict__ inv, int n) {
  int i = blockIdx.x * blockDim.x + threadIdx.x;
  if (i < n) inv[i] = 1.0f / fmaxf((float)cnt[i], 1.0f);
}

// One block per edge; threads stride over channels. f32 atomics resolve in L2.
__global__ void scatter_add_kernel(const float* __restrict__ x,
                                   const int* __restrict__ src,
                                   const int* __restrict__ dst,
                                   float* __restrict__ agg, int C) {
  int e = blockIdx.x;
  int s = src[e], d = dst[e];
  const float* xr = x + (size_t)s * C;
  float*       ar = agg + (size_t)d * C;
  for (int c = threadIdx.x; c < C; c += blockDim.x)
    atomicAdd(&ar[c], xr[c]);
}

__global__ void log_softmax_kernel(const float* __restrict__ in,
                                   float* __restrict__ out, int rows, int ncol) {
  int r = blockIdx.x * blockDim.x + threadIdx.x;
  if (r >= rows) return;
  const float* p = in + (size_t)r * ncol;
  float m = -INFINITY;
  for (int c = 0; c < ncol; ++c) m = fmaxf(m, p[c]);
  float sum = 0.0f;
  for (int c = 0; c < ncol; ++c) sum += expf(p[c] - m);
  float lse = m + logf(sum);
  float* q = out + (size_t)r * ncol;
  for (int c = 0; c < ncol; ++c) q[c] = p[c] - lse;
}

// ---------------------------------------------------------------------------
// Fragment packing (ISA 7.12.2 wave32 layouts, baked into memory):
//   A (16x32 bf16): lane h*16+l -> row l; elem j -> K kmap(j,h).
//   B (32x16 bf16): lane -> col l; elem j -> K kmap(j,h) (same as A).
// Packed storage: frag(tile, chunk) = 32 lanes x 16 bf16, lane-major:
//   A: [mTile][chunk][lane][16]   B: [colTile16][chunk][lane][16]
// 'chunk' ranges over the K-concatenation of both GEMM streams.
// ---------------------------------------------------------------------------
__global__ void pack_a_kernel(const float* __restrict__ A,
                              const float* __restrict__ scale,  // nullable
                              __bf16* __restrict__ out,
                              int tilesM, int K, int lda,
                              int nChTot, int chOff, int nChSrc) {
  int idx = blockIdx.x * blockDim.x + threadIdx.x;
  if (idx >= tilesM * nChSrc * 32) return;
  int lane = idx & 31;
  int q    = (idx >> 5) % nChSrc;
  int t    = (idx >> 5) / nChSrc;
  int h = lane >> 4, l = lane & 15;
  int row = t * 16 + l;                       // M is always a multiple of 16
  const float* ap = A + (size_t)row * lda;
  float sc = scale ? scale[row] : 1.0f;
  int k0 = q * 32;
  v16bf v;
#pragma unroll
  for (int j = 0; j < 16; ++j) {
    int vv = j >> 1;
    int kk = ((vv < 4) ? (8 * h + 2 * vv) : (16 + 8 * h + 2 * (vv - 4))) + (j & 1);
    int k  = k0 + kk;
    float f = (k < K) ? ap[k] * sc : 0.0f;    // zero-pad ragged K at pack time
    v[j] = (__bf16)f;
  }
  *(v16bf*)(out + ((size_t)(t * nChTot + chOff + q) * 32 + lane) * 16) = v;
}

__global__ void pack_b_kernel(const float* __restrict__ W,
                              __bf16* __restrict__ out,
                              int nCT, int K, int N,
                              int nChTot, int chOff, int nChSrc) {
  int idx = blockIdx.x * blockDim.x + threadIdx.x;
  if (idx >= nCT * nChSrc * 32) return;
  int lane = idx & 31;
  int q    = (idx >> 5) % nChSrc;
  int ct   = (idx >> 5) / nChSrc;
  int h = lane >> 4, l = lane & 15;
  int col = ct * 16 + l;
  int k0 = q * 32;
  v16bf v;
#pragma unroll
  for (int j = 0; j < 16; ++j) {
    int k = k0 + ((j < 8) ? (8 * h + j) : (16 + 8 * h + (j - 8)));
    float f = (k < K && col < N) ? W[(size_t)k * N + col] : 0.0f;  // pad K and N
    v[j] = (__bf16)f;
  }
  *(v16bf*)(out + ((size_t)(ct * nChTot + chOff + q) * 32 + lane) * 16) = v;
}

// ---------------------------------------------------------------------------
// Packed GEMM: C[M,N] = Apk @ Bpk + bias (+relu).
// One wave owns a 32x64 strip: per chunk 2 A-frags + 4 B-frags -> 8 wmma.
// All intra-chunk addresses are immediate offsets from two moving pointers.
// ---------------------------------------------------------------------------
#define GEMM_BLOCK 256

__global__ void __launch_bounds__(GEMM_BLOCK)
sage_gemm_packed(const __bf16* __restrict__ Apk, const __bf16* __restrict__ Bpk,
                 const float* __restrict__ bias, float* __restrict__ Cout,
                 int mPairs, int tilesN, int nCh, int N, int ldc, int relu) {
  int wave = blockIdx.x * (GEMM_BLOCK / 32) + (threadIdx.x >> 5);
  if (wave >= mPairs * tilesN) return;       // uniform per wave: EXEC all-ones

  int tp = wave / tilesN;                    // M-tile pair (rows tp*32..+31)
  int tn = wave % tilesN;
  int lane = threadIdx.x & 31;

  const size_t tileOff = (size_t)nCh * 512;  // elements between adjacent tiles
  const __bf16* ap = Apk + ((size_t)(tp * 2) * nCh * 32 + lane) * 16;
  const __bf16* bp = Bpk + ((size_t)(tn * 4) * nCh * 32 + lane) * 16;

  v8f acc0[4] = {v8f{}, v8f{}, v8f{}, v8f{}};
  v8f acc1[4] = {v8f{}, v8f{}, v8f{}, v8f{}};

#pragma unroll 2
  for (int q = 0; q < nCh; ++q) {
    v16bf a0 = *(const v16bf*)ap;                       // 2x b128
    v16bf a1 = *(const v16bf*)(ap + tileOff);           // imm offset
#pragma unroll
    for (int nn = 0; nn < 4; ++nn) {
      v16bf b = *(const v16bf*)(bp + (size_t)nn * tileOff);  // imm offsets
      acc0[nn] = __builtin_amdgcn_wmma_f32_16x16x32_bf16(
          false, a0, false, b, (short)0, acc0[nn], false, false);
      acc1[nn] = __builtin_amdgcn_wmma_f32_16x16x32_bf16(
          false, a1, false, b, (short)0, acc1[nn], false, false);
    }
    ap += 512;
    bp += 512;
  }

  int h = lane >> 4, l = lane & 15;
  int m0 = tp << 5, n0 = tn << 6;
#pragma unroll
  for (int nn = 0; nn < 4; ++nn) {
    int col = n0 + nn * 16 + l;
    if (col < N) {
      float bcol = bias[col];
#pragma unroll
      for (int v = 0; v < 8; ++v) {
        int row0 = m0 + v + 8 * h;
        float o0 = acc0[nn][v] + bcol;
        float o1 = acc1[nn][v] + bcol;
        if (relu) { o0 = fmaxf(o0, 0.0f); o1 = fmaxf(o1, 0.0f); }
        Cout[(size_t)row0 * ldc + col] = o0;
        Cout[(size_t)(row0 + 16) * ldc + col] = o1;
      }
    }
  }
}

// ---------------------------------------------------------------------------
// Host-side orchestration
// ---------------------------------------------------------------------------
static inline void zero_words(void* p, long long n, hipStream_t s) {
  zero_u32_kernel<<<(int)((n + 255) / 256), 256, 0, s>>>((unsigned int*)p, n);
}

extern "C" void kernel_launch(void* const* d_in, const int* in_sizes, int n_in,
                              void* d_out, int out_size, void* d_ws, size_t ws_size,
                              hipStream_t stream) {
  // ---- inputs (reference setup_inputs order) ----
  const float* x    = (const float*)d_in[0];    // (400000,100)
  const int*   src0 = (const int*)d_in[1];
  const int*   dst0 = (const int*)d_in[2];
  const int*   src1 = (const int*)d_in[3];
  const int*   dst1 = (const int*)d_in[4];
  const int*   src2 = (const int*)d_in[5];
  const int*   dst2 = (const int*)d_in[6];
  const float* Wl[3] = {(const float*)d_in[7],  (const float*)d_in[10], (const float*)d_in[13]};
  const float* bl[3] = {(const float*)d_in[8],  (const float*)d_in[11], (const float*)d_in[14]};
  const float* Wr[3] = {(const float*)d_in[9],  (const float*)d_in[12], (const float*)d_in[15]};

  const int E[3]    = {in_sizes[1], in_sizes[3], in_sizes[5]};
  const int NT[3]   = {100000, 20000, 4096};     // n_tgt per layer
  const int CIN[3]  = {100, 1024, 1024};
  const int COUT[3] = {1024, 1024, 47};

  // ---- outputs: [log_softmax 4096x47 | logits 4096x47 | out_saved 20000x1024]
  float* out_lsm    = (float*)d_out;
  float* out_logits = out_lsm + (size_t)4096 * 47;
  float* out_saved  = out_logits + (size_t)4096 * 47;   // layer-1 relu output

  // ---- workspace layout (all offsets 32B-aligned) ----
  float*  h1  = (float*)d_ws;                            // 100000*1024 f32
  float*  agg = h1 + (size_t)100000 * 1024;              // up to 20000*1024 f32
  int*    cnt = (int*)(agg + (size_t)20000 * 1024);      // up to 100000 i32
  float*  inv = (float*)(cnt + 100000);                  // up to 100000 f32
  __bf16* Apk = (__bf16*)(inv + 100000);                 // up to 40,960,000 bf16
  __bf16* Bpk = Apk + (size_t)40960000;                  // up to  2,097,152 bf16

  const float* layerIn[3]  = {x, h1, out_saved};
  float*       layerOut[3] = {h1, out_saved, out_logits};

  for (int i = 0; i < 3; ++i) {
    const int nT = NT[i], K = CIN[i], N = COUT[i];
    const int tilesM = nT / 16;                  // 6250 / 1250 / 256: all even
    const int mPairs = tilesM / 2;
    const int tilesN = (N + 63) / 64;
    const int nCT    = tilesN * 4;               // 16-col tiles (zero-padded)
    const int nCh1   = (K + 31) / 32;            // chunks per stream
    const int nChTot = 2 * nCh1;                 // agg-stream ++ x_tgt-stream

    // -- mean aggregation --
    zero_words(cnt, nT, stream);
    zero_words(agg, (long long)nT * K, stream);
    count_kernel<<<(E[i] + 255) / 256, 256, 0, stream>>>(
        (i == 0 ? dst0 : i == 1 ? dst1 : dst2), cnt, E[i]);
    invdeg_kernel<<<(nT + 255) / 256, 256, 0, stream>>>(cnt, inv, nT);
    scatter_add_kernel<<<E[i], (K < 256 ? 128 : 256), 0, stream>>>(
        layerIn[i], (i == 0 ? src0 : i == 1 ? src1 : src2),
        (i == 0 ? dst0 : i == 1 ? dst1 : dst2), agg, K);

    // -- pack operands into WMMA fragment layout (bf16, padded) --
    {
      int thA = tilesM * nCh1 * 32;
      pack_a_kernel<<<(thA + 255) / 256, 256, 0, stream>>>(
          agg, inv, Apk, tilesM, K, K, nChTot, 0, nCh1);        // scaled agg
      pack_a_kernel<<<(thA + 255) / 256, 256, 0, stream>>>(
          layerIn[i], nullptr, Apk, tilesM, K, K, nChTot, nCh1, nCh1);  // x_tgt
      int thB = nCT * nCh1 * 32;
      pack_b_kernel<<<(thB + 255) / 256, 256, 0, stream>>>(
          Wl[i], Bpk, nCT, K, N, nChTot, 0, nCh1);
      pack_b_kernel<<<(thB + 255) / 256, 256, 0, stream>>>(
          Wr[i], Bpk, nCT, K, N, nChTot, nCh1, nCh1);
    }

    // -- WMMA GEMM + bias (+relu) --
    {
      long long tiles = (long long)mPairs * tilesN;
      int blocks = (int)((tiles + (GEMM_BLOCK / 32) - 1) / (GEMM_BLOCK / 32));
      sage_gemm_packed<<<blocks, GEMM_BLOCK, 0, stream>>>(
          Apk, Bpk, bl[i], layerOut[i], mPairs, tilesN, nChTot, N, N,
          /*relu=*/(i != 2));
    }
  }

  // ---- log_softmax over final logits ----
  log_softmax_kernel<<<(4096 + 255) / 256, 256, 0, stream>>>(out_logits, out_lsm,
                                                             4096, 47);
}
